// TransNextBlock_20212116095547
// MI455X (gfx1250) — matrix operand order repaired and checked by reading, not compiled, not run on hardware
//
#include <hip/hip_runtime.h>

// ---------------- problem constants ----------------
#define B_   2
#define C_   128
#define H_   56
#define W_   56
#define N_   3136           // H_*W_
#define NH_  4
#define D_   32             // head dim
#define TOK  (B_*N_)        // 6272 tokens
#define HF_  341            // hidden of conv-GLU
#define F1P  704            // 2*HF_=682 padded to 64 (44 WMMA tiles = 11 groups of 4)
#define F2KP 352            // HF_=341 padded to 32

typedef _Float16 v16h __attribute__((ext_vector_type(16)));
typedef _Float16 v8h  __attribute__((ext_vector_type(8)));
typedef float    v8f  __attribute__((ext_vector_type(8)));

__device__ __forceinline__ float gelu_exact(float x) {
    return 0.5f * x * (1.0f + erff(x * 0.7071067811865476f));
}

__device__ __forceinline__ v8f wmma_f16(v16h a, v16h b, v8f c) {
    return __builtin_amdgcn_wmma_f32_16x16x32_f16(false, a, false, b, (short)0, c, false, false);
}

// A fragment (16x32 f16, M x K): lane m=l&15, hi=l>>4:
//   halves[0..7]=row[hi*8 + j], halves[8..15]=row[16 + hi*8 + j]
__device__ __forceinline__ v16h load_afrag(const _Float16* row, int hi) {
    v8h lo = *(const v8h*)(row + hi * 8);
    v8h hp = *(const v8h*)(row + 16 + hi * 8);
    return __builtin_shufflevector(lo, hp, 0,1,2,3,4,5,6,7,8,9,10,11,12,13,14,15);
}

// B fragment (32x16 f16, K x N): lane n=l&15, hi=l>>4:
//   halves[0..7]=col_n[hi*8 + j], halves[8..15]=col_n[16 + hi*8 + j]
__device__ __forceinline__ v16h load_bfrag(const _Float16* col, int hi) {
    const _Float16* p = col + hi * 8;
    v8h lo = *(const v8h*)(p);
    v8h hp = *(const v8h*)(p + 16);
    return __builtin_shufflevector(lo, hp, 0,1,2,3,4,5,6,7,8,9,10,11,12,13,14,15);
}

// ---------------- weight convert + pad to f16 ----------------
__global__ void cvt_pad(const float* __restrict__ src, _Float16* __restrict__ dst,
                        int Oreal, int Kreal, int Kpad, int Opad) {
    int i = blockIdx.x * blockDim.x + threadIdx.x;
    if (i >= Opad * Kpad) return;
    int o = i / Kpad, k = i - o * Kpad;
    float v = (o < Oreal && k < Kreal) ? src[o * Kreal + k] : 0.0f;
    dst[i] = (_Float16)v;
}

// ---------------- LayerNorm over C per token (optional gelu pre-op) ----------------
// chanMajor: input indexed (b*C+c)*N + n ; else token-major tok*C + c
__global__ void ln_kernel(const float* __restrict__ in, const float* __restrict__ w,
                          const float* __restrict__ bvec, float* __restrict__ outf,
                          _Float16* __restrict__ outh, int chanMajor, int doGelu) {
    int tok = blockIdx.x;
    int c = threadIdx.x;
    int b = tok / N_;
    int n = tok - b * N_;
    size_t idx = chanMajor ? ((size_t)(b * C_ + c)) * N_ + n : (size_t)tok * C_ + c;
    float x = in[idx];
    if (doGelu) x = gelu_exact(x);
    __shared__ float s1[C_], s2[C_];
    s1[c] = x; s2[c] = x * x;
    __syncthreads();
    for (int st = 64; st > 0; st >>= 1) {
        if (c < st) { s1[c] += s1[c + st]; s2[c] += s2[c + st]; }
        __syncthreads();
    }
    float mean = s1[0] * (1.0f / C_);
    float var  = s2[0] * (1.0f / C_) - mean * mean;
    float inv  = rsqrtf(var + 1e-5f);
    float o = (x - mean) * inv * w[c] + bvec[c];
    size_t oidx = (size_t)tok * C_ + c;
    if (outf) outf[oidx] = o;
    if (outh) outh[oidx] = (_Float16)o;
}

// ---------------- WMMA GEMM, 16x64 per wave (static 4-tile unroll, no dyn VGPR idx) -------
// out[M x Opad] = A[M x K] * W[Opad x K]^T + bias ; requires Opad % 64 == 0
__global__ void gemm_wmma(const _Float16* __restrict__ A, const _Float16* __restrict__ Wt,
                          const float* __restrict__ bias, float* __restrict__ out,
                          int M, int K, int Opad, int Oreal) {
    int groupsN = Opad >> 6;                 // 64-wide column groups
    int wave = blockIdx.x * (blockDim.x >> 5) + (threadIdx.x >> 5);
    int tm = wave / groupsN;
    int tg = wave - tm * groupsN;
    if (tm * 16 >= M) return;
    int lane = threadIdx.x & 31;
    int r = lane & 15, hi = lane >> 4;
    const _Float16* ap  = A  + (size_t)(tm * 16 + r) * K;
    int tn0 = tg * 4;
    const _Float16* wp0 = Wt + (size_t)(tn0 * 16 + r) * K;
    v8f a0 = {}, a1 = {}, a2 = {}, a3 = {};
    for (int k0 = 0; k0 < K; k0 += 32) {
        v16h av = load_afrag(ap + k0, hi);
        if (k0 + 32 < K) __builtin_prefetch(ap + k0 + 32, 0, 3);   // WGP-scope prefetch
        a0 = wmma_f16(av, load_bfrag(wp0 + k0, hi), a0);
        a1 = wmma_f16(av, load_bfrag(wp0 + (size_t)16 * K + k0, hi), a1);
        a2 = wmma_f16(av, load_bfrag(wp0 + (size_t)32 * K + k0, hi), a2);
        a3 = wmma_f16(av, load_bfrag(wp0 + (size_t)48 * K + k0, hi), a3);
    }
    int rowb = tm * 16 + hi * 8;
#define STORE_TILE(J, ACC)                                                      \
    {                                                                           \
        int ocol = (tn0 + (J)) * 16 + r;                                        \
        float bb = (bias && ocol < Oreal) ? bias[ocol] : 0.0f;                  \
        _Pragma("unroll")                                                       \
        for (int v = 0; v < 8; v++)                                             \
            out[(size_t)(rowb + v) * Opad + ocol] = ACC[v] + bb;                \
    }
    STORE_TILE(0, a0)
    STORE_TILE(1, a1)
    STORE_TILE(2, a2)
    STORE_TILE(3, a3)
#undef STORE_TILE
}

// ---------------- q -> qn (f32), qs (f32,f16) ----------------
__global__ void q_prep(const float* __restrict__ qf, const float* __restrict__ temp,
                       const float* __restrict__ qe, float* __restrict__ qn,
                       float* __restrict__ qsf, _Float16* __restrict__ qsh) {
    int i = blockIdx.x * blockDim.x + threadIdx.x;
    if (i >= B_ * NH_ * N_) return;
    int n = i % N_; int bh = i / N_; int b = bh >> 2; int h = bh & 3;
    const float* src = qf + ((size_t)b * N_ + n) * C_ + h * D_;
    float v[D_]; float ss = 0.0f;
    #pragma unroll
    for (int d = 0; d < D_; d++) { v[d] = src[d]; ss += v[d] * v[d]; }
    float inv = 1.0f / fmaxf(sqrtf(ss), 1e-12f);
    float sp = log1pf(__expf(temp[h]));
    size_t o = (size_t)i * D_;
    #pragma unroll
    for (int d = 0; d < D_; d++) {
        float qv = v[d] * inv;
        qn[o + d] = qv;
        float qs = (qv + qe[h * D_ + d]) * sp;
        qsf[o + d] = qs; qsh[o + d] = (_Float16)qs;
    }
}

// ---------------- kv -> kvl (B, 2C, N) with k l2-normalized per (n,h) ----------------
__global__ void kvl_prep(const float* __restrict__ kv, float* __restrict__ kvl) {
    int i = blockIdx.x * blockDim.x + threadIdx.x;
    if (i >= B_ * NH_ * N_) return;
    int n = i % N_; int bh = i / N_; int b = bh >> 2; int h = bh & 3;
    const float* src = kv + ((size_t)b * N_ + n) * (2 * C_) + h * D_;
    float v[D_]; float ss = 0.0f;
    #pragma unroll
    for (int d = 0; d < D_; d++) { v[d] = src[d]; ss += v[d] * v[d]; }
    float inv = 1.0f / fmaxf(sqrtf(ss), 1e-12f);
    float* kdst = kvl + ((size_t)b * 2 * C_ + h * D_) * N_ + n;
    #pragma unroll
    for (int d = 0; d < D_; d++) kdst[(size_t)d * N_] = v[d] * inv;
    const float* vs = src + C_;
    float* vdst = kvl + ((size_t)b * 2 * C_ + C_ + h * D_) * N_ + n;
    #pragma unroll
    for (int d = 0; d < D_; d++) vdst[(size_t)d * N_] = vs[d];
}

// ---------------- local 3x3 logits: attnl[b,h,n,9] (zero padding => logit 0) ----------------
__global__ void local_logits(const float* __restrict__ qsf, const float* __restrict__ kvl,
                             float* __restrict__ attnl) {
    int i = blockIdx.x * blockDim.x + threadIdx.x;
    if (i >= B_ * NH_ * N_) return;
    int n = i % N_; int bh = i / N_; int b = bh >> 2; int h = bh & 3;
    int iy = n / W_, ix = n - iy * W_;
    const float* qp = qsf + (size_t)i * D_;
    const float* kbase = kvl + ((size_t)b * 2 * C_ + h * D_) * N_;
    float* op = attnl + (size_t)i * 9;
    for (int kk = 0; kk < 9; kk++) {
        int yy = iy + kk / 3 - 1, xx = ix + kk % 3 - 1;
        float dot = 0.0f;
        if (yy >= 0 && yy < H_ && xx >= 0 && xx < W_) {
            int pos = yy * W_ + xx;
            #pragma unroll
            for (int d = 0; d < D_; d++) dot += qp[d] * kbase[(size_t)d * N_ + pos];
        }
        op[kk] = dot;
    }
}

// ---------------- kvp -> kph (b,h,n,d f16 normalized), vpht (b,h,d,n f16) ----------------
__global__ void kvp_prep(const float* __restrict__ kvp, _Float16* __restrict__ kph,
                         _Float16* __restrict__ vpht) {
    int i = blockIdx.x * blockDim.x + threadIdx.x;
    if (i >= B_ * NH_ * N_) return;
    int n = i % N_; int bh = i / N_; int b = bh >> 2; int h = bh & 3;
    const float* src = kvp + ((size_t)b * N_ + n) * (2 * C_) + h * D_;
    float v[D_]; float ss = 0.0f;
    #pragma unroll
    for (int d = 0; d < D_; d++) { v[d] = src[d]; ss += v[d] * v[d]; }
    float inv = 1.0f / fmaxf(sqrtf(ss), 1e-12f);
    size_t ko = (size_t)i * D_;
    #pragma unroll
    for (int d = 0; d < D_; d++) kph[ko + d] = (_Float16)(v[d] * inv);
    const float* vs = src + C_;
    _Float16* vd = vpht + ((size_t)bh * D_) * N_ + n;
    #pragma unroll
    for (int d = 0; d < D_; d++) vd[(size_t)d * N_] = (_Float16)vs[d];
}

// ---------------- flash pooled attention, LDS-staged K/V shared by 4 waves ----------------
// one wave = (b,h, 16-query tile); 4 waves per block share each 32-key K/V tile via LDS
#define FP_WAVES 4
__global__ void flash_pool(const _Float16* __restrict__ qsh, const _Float16* __restrict__ kph,
                           const _Float16* __restrict__ vpht, const float* __restrict__ attnl,
                           float* __restrict__ xattn, float* __restrict__ mbuf,
                           float* __restrict__ zbuf) {
    const int QT = N_ / 16;              // 196 query tiles per (b,h)
    const int GB = QT / FP_WAVES;        // 49 blocks per (b,h)
    int bh = blockIdx.x / GB;
    int qt = (blockIdx.x % GB) * FP_WAVES + (threadIdx.x >> 5);
    int b = bh >> 2; int h = bh & 3;
    int lane = threadIdx.x & 31;
    int q = lane & 15; int hi = lane >> 4;
    int qrow = qt * 16 + q;
    size_t bhN = (size_t)bh * N_;
    const _Float16* kbase = kph  + bhN * D_;     // (key, d) row-major
    const _Float16* vbase = vpht + bhN * D_;     // (d, key) row-major

    __shared__ __align__(16) _Float16 sk[2][32 * 32];  // [key][d]
    __shared__ __align__(16) _Float16 sv[2][32 * 32];  // [d][key]
    int tid = threadIdx.x;
    int lrow = tid >> 2, lcol = (tid & 3) * 8;   // 128 threads x 16B covers a 2KB tile

    // B fragment: Q^T  (B[k][n]=Q[n][k]) -> contiguous row of qs
    v16h bq = load_bfrag(qsh + (bhN + qrow) * D_, hi);
    // seed online softmax with the 9 local logits (exact concat-softmax)
    float m_run = -1e30f, Z = 0.0f;
    const float* lp = attnl + (bhN + qrow) * 9;
    #pragma unroll
    for (int k = 0; k < 9; k++) m_run = fmaxf(m_run, lp[k]);
    #pragma unroll
    for (int k = 0; k < 9; k++) Z += __expf(lp[k] - m_run);
    v8f accA = {}, accB = {};

    // preload tile 0
    *(v8h*)&sk[0][lrow * 32 + lcol] = *(const v8h*)(kbase + (size_t)lrow * D_ + lcol);
    *(v8h*)&sv[0][lrow * 32 + lcol] = *(const v8h*)(vbase + (size_t)lrow * N_ + lcol);

    const int NT = N_ / 32;              // 98 key tiles
    for (int jt = 0; jt < NT; jt++) {
        __syncthreads();
        int buf = jt & 1;
        if (jt + 1 < NT) {               // double-buffered staging
            int j0n = (jt + 1) * 32;
            *(v8h*)&sk[buf ^ 1][lrow * 32 + lcol] =
                *(const v8h*)(kbase + (size_t)(j0n + lrow) * D_ + lcol);
            *(v8h*)&sv[buf ^ 1][lrow * 32 + lcol] =
                *(const v8h*)(vbase + (size_t)lrow * N_ + j0n + lcol);
            if (jt + 2 < NT) {
                __builtin_prefetch(kbase + (size_t)(j0n + 32 + lrow) * D_ + lcol, 0, 3);
                __builtin_prefetch(vbase + (size_t)lrow * N_ + j0n + 32 + lcol, 0, 3);
            }
        }
        v16h ka1 = load_afrag(&sk[buf][q * 32], hi);
        v16h ka2 = load_afrag(&sk[buf][(16 + q) * 32], hi);
        v8f zero8 = {};
        v8f s1 = wmma_f16(ka1, bq, zero8);   // S^T[key][q]
        v8f s2 = wmma_f16(ka2, bq, zero8);
        float cm = -1e30f;
        #pragma unroll
        for (int v = 0; v < 8; v++) cm = fmaxf(cm, fmaxf(s1[v], s2[v]));
        cm = fmaxf(cm, __shfl_xor(cm, 16, 32));
        float mnew = fmaxf(m_run, cm);
        float corr = __expf(m_run - mnew);
        float p1[8], p2[8]; float rs = 0.0f;
        #pragma unroll
        for (int v = 0; v < 8; v++) {
            p1[v] = __expf(s1[v] - mnew);
            p2[v] = __expf(s2[v] - mnew);
            rs += p1[v] + p2[v];
        }
        rs += __shfl_xor(rs, 16, 32);
        Z = Z * corr + rs;
        m_run = mnew;
        #pragma unroll
        for (int v = 0; v < 8; v++) {        // rescale accumulators per query row
            float f = __shfl(corr, v + hi * 8, 32);
            accA[v] *= f; accB[v] *= f;
        }
        // P^T D-layout is exactly the A-fragment layout for P (zero shuffles):
        v16h pf;
        #pragma unroll
        for (int v = 0; v < 8; v++) { pf[v] = (_Float16)p1[v]; pf[8 + v] = (_Float16)p2[v]; }
        v16h vfA = load_bfrag(&sv[buf][q * 32], hi);          // dd = q (0..15)
        v16h vfB = load_bfrag(&sv[buf][(16 + q) * 32], hi);   // dd = 16+q
        accA = wmma_f16(pf, vfA, accA);
        accB = wmma_f16(pf, vfB, accB);
    }
    if (lane < 16) { mbuf[bhN + qrow] = m_run; zbuf[bhN + qrow] = Z; }
    #pragma unroll
    for (int v = 0; v < 8; v++) {
        int row = qt * 16 + v + hi * 8;
        float zr = __shfl(Z, v + hi * 8, 32);
        float invz = 1.0f / zr;
        size_t o = ((size_t)b * N_ + row) * C_ + h * D_;
        xattn[o + q]      = accA[v] * invz;
        xattn[o + 16 + q] = accB[v] * invz;
    }
}

// ---------------- local branch: xattn += (qn.tokens + bias + a_l) . v_local ----------------
__global__ void local_apply(const float* __restrict__ qn, const float* __restrict__ lt,
                            const float* __restrict__ lb, const float* __restrict__ attnl,
                            const float* __restrict__ kvl, const float* __restrict__ mbuf,
                            const float* __restrict__ zbuf, float* __restrict__ xattn,
                            _Float16* __restrict__ xattnh) {
    int i = blockIdx.x * blockDim.x + threadIdx.x;
    if (i >= B_ * NH_ * N_) return;
    int n = i % N_; int bh = i / N_; int b = bh >> 2; int h = bh & 3;
    int iy = n / W_, ix = n - iy * W_;
    float m = mbuf[i], invz = 1.0f / zbuf[i];
    const float* lp = attnl + (size_t)i * 9;
    const float* qp = qn + (size_t)i * D_;
    float wl[9]; int pos[9]; bool ok[9];
    for (int kk = 0; kk < 9; kk++) {
        float s = lb[h * 9 + kk];
        const float* tp = lt + ((size_t)h * D_) * 9 + kk;   // learnable_tokens (nh,d,9)
        #pragma unroll
        for (int d = 0; d < D_; d++) s += qp[d] * tp[d * 9];
        wl[kk] = s + __expf(lp[kk] - m) * invz;
        int yy = iy + kk / 3 - 1, xx = ix + kk % 3 - 1;
        ok[kk] = (yy >= 0 && yy < H_ && xx >= 0 && xx < W_);
        pos[kk] = yy * W_ + xx;
    }
    const float* vb = kvl + ((size_t)b * 2 * C_ + C_ + h * D_) * N_;
    for (int d = 0; d < D_; d++) {
        float acc = 0.0f;
        #pragma unroll
        for (int kk = 0; kk < 9; kk++)
            if (ok[kk]) acc += wl[kk] * vb[(size_t)d * N_ + pos[kk]];
        size_t o = ((size_t)b * N_ + n) * C_ + h * D_ + d;
        float val = xattn[o] + acc;
        xattn[o] = val; xattnh[o] = (_Float16)val;
    }
}

// ---------------- residual add with layout transpose (B,C,N) += (B,N,C) ----------------
__global__ void add_res(const float* __restrict__ x, const float* __restrict__ tokmaj,
                        float* __restrict__ out) {
    int i = blockIdx.x * blockDim.x + threadIdx.x;
    if (i >= TOK * C_) return;
    int n = i % N_; int bc = i / N_; int b = bc / C_; int c = bc - b * C_;
    out[i] = x[i] + tokmaj[((size_t)b * N_ + n) * C_ + c];
}

// ---------------- depthwise 3x3 + GLU gate -> g (f16, K padded to 352) ----------------
__global__ void dwglu(const float* __restrict__ f1, const float* __restrict__ dww,
                      const float* __restrict__ dwb, _Float16* __restrict__ gh) {
    int i = blockIdx.x * blockDim.x + threadIdx.x;
    if (i >= TOK * F2KP) return;
    int ch = i % F2KP; int t = i / F2KP; int n = t % N_; int b = t / N_;
    size_t oidx = (size_t)t * F2KP + ch;
    if (ch >= HF_) { gh[oidx] = (_Float16)0.0f; return; }
    int iy = n / W_, ix = n - iy * W_;
    float acc = dwb[ch];
    for (int kk = 0; kk < 9; kk++) {
        int yy = iy + kk / 3 - 1, xx = ix + kk % 3 - 1;
        if (yy >= 0 && yy < H_ && xx >= 0 && xx < W_)
            acc += f1[((size_t)b * N_ + yy * W_ + xx) * F1P + ch] * dww[ch * 9 + kk];
    }
    float vg = f1[(size_t)t * F1P + HF_ + ch];
    gh[oidx] = (_Float16)(gelu_exact(acc) * vg);
}

// ---------------- final: out(B,C,N) = x1(B,C,N) + mlp(B,N,C) ----------------
__global__ void final_add(const float* __restrict__ x1, const float* __restrict__ mlp,
                          float* __restrict__ out) {
    int i = blockIdx.x * blockDim.x + threadIdx.x;
    if (i >= TOK * C_) return;
    int n = i % N_; int bc = i / N_; int b = bc / C_; int c = bc - b * C_;
    out[i] = x1[i] + mlp[((size_t)b * N_ + n) * C_ + c];
}

// =======================================================================================
extern "C" void kernel_launch(void* const* d_in, const int* in_sizes, int n_in,
                              void* d_out, int out_size, void* d_ws, size_t ws_size,
                              hipStream_t stream) {
    (void)in_sizes; (void)n_in; (void)out_size; (void)ws_size;
    const float* x       = (const float*)d_in[0];
    const float* norm1_w = (const float*)d_in[1];
    const float* norm1_b = (const float*)d_in[2];
    const float* q_w     = (const float*)d_in[3];
    const float* q_b     = (const float*)d_in[4];
    const float* kv_w    = (const float*)d_in[5];
    const float* kv_b    = (const float*)d_in[6];
    const float* temp    = (const float*)d_in[7];
    const float* qemb    = (const float*)d_in[8];
    const float* ltok    = (const float*)d_in[9];
    const float* lbias   = (const float*)d_in[10];
    const float* sr_w    = (const float*)d_in[11];
    const float* sr_b    = (const float*)d_in[12];
    const float* pln_w   = (const float*)d_in[13];
    const float* pln_b   = (const float*)d_in[14];
    const float* proj_w  = (const float*)d_in[15];
    const float* proj_b  = (const float*)d_in[16];
    const float* norm2_w = (const float*)d_in[17];
    const float* norm2_b = (const float*)d_in[18];
    const float* fc1_w   = (const float*)d_in[19];
    const float* fc1_b   = (const float*)d_in[20];
    const float* dw_w    = (const float*)d_in[21];
    const float* dw_b    = (const float*)d_in[22];
    const float* fc2_w   = (const float*)d_in[23];
    const float* fc2_b   = (const float*)d_in[24];

    char* ws = (char*)d_ws; size_t off = 0;
    auto alloc = [&](size_t bytes) -> void* {
        size_t a = (off + 255) & ~(size_t)255; off = a + bytes; return ws + a;
    };
    _Float16* y_h    = (_Float16*)alloc((size_t)TOK * C_ * 2);
    _Float16* qwh    = (_Float16*)alloc((size_t)128 * 128 * 2);
    _Float16* kvwh   = (_Float16*)alloc((size_t)256 * 128 * 2);
    _Float16* srwh   = (_Float16*)alloc((size_t)128 * 128 * 2);
    _Float16* projwh = (_Float16*)alloc((size_t)128 * 128 * 2);
    _Float16* fc1wh  = (_Float16*)alloc((size_t)F1P * 128 * 2);
    _Float16* fc2wh  = (_Float16*)alloc((size_t)128 * F2KP * 2);
    float*    q_f    = (float*)alloc((size_t)TOK * 128 * 4);
    float*    kv_f   = (float*)alloc((size_t)TOK * 256 * 4);
    float*    qn_f   = (float*)alloc((size_t)TOK * 128 * 4);
    float*    qs_f   = (float*)alloc((size_t)TOK * 128 * 4);
    _Float16* qs_h   = (_Float16*)alloc((size_t)TOK * 128 * 2);
    float*    kvl_f  = (float*)alloc((size_t)TOK * 256 * 4);
    float*    attnl  = (float*)alloc((size_t)B_ * NH_ * N_ * 9 * 4);
    float*    s_f    = (float*)alloc((size_t)TOK * 128 * 4);
    _Float16* xp_h   = (_Float16*)alloc((size_t)TOK * 128 * 2);
    float*    kvp_f  = (float*)alloc((size_t)TOK * 256 * 4);
    _Float16* kph    = (_Float16*)alloc((size_t)TOK * 128 * 2);
    _Float16* vpht   = (_Float16*)alloc((size_t)TOK * 128 * 2);
    float*    mbuf   = (float*)alloc((size_t)B_ * NH_ * N_ * 4);
    float*    zbuf   = (float*)alloc((size_t)B_ * NH_ * N_ * 4);
    float*    xat_f  = (float*)alloc((size_t)TOK * 128 * 4);
    _Float16* xat_h  = (_Float16*)alloc((size_t)TOK * 128 * 2);
    float*    proj_f = (float*)alloc((size_t)TOK * 128 * 4);
    float*    x1_f   = (float*)alloc((size_t)TOK * 128 * 4);
    _Float16* y2_h   = (_Float16*)alloc((size_t)TOK * 128 * 2);
    float*    f1_f   = (float*)alloc((size_t)TOK * F1P * 4);
    _Float16* g_h    = (_Float16*)alloc((size_t)TOK * F2KP * 2);
    float*    mlp_f  = (float*)alloc((size_t)TOK * 128 * 4);

    const int TPB = 256;
    auto grid1 = [&](long n) { return dim3((unsigned)((n + TPB - 1) / TPB)); };
    // weights -> f16 (with padding where needed)
    cvt_pad<<<grid1(128L * 128), TPB, 0, stream>>>(q_w, qwh, 128, 128, 128, 128);
    cvt_pad<<<grid1(256L * 128), TPB, 0, stream>>>(kv_w, kvwh, 256, 128, 128, 256);
    cvt_pad<<<grid1(128L * 128), TPB, 0, stream>>>(sr_w, srwh, 128, 128, 128, 128);
    cvt_pad<<<grid1(128L * 128), TPB, 0, stream>>>(proj_w, projwh, 128, 128, 128, 128);
    cvt_pad<<<grid1((long)F1P * 128), TPB, 0, stream>>>(fc1_w, fc1wh, 682, 128, 128, F1P);
    cvt_pad<<<grid1(128L * F2KP), TPB, 0, stream>>>(fc2_w, fc2wh, 128, HF_, F2KP, 128);

    auto gemm = [&](const _Float16* A, const _Float16* Wt, const float* bias, float* out,
                    int M, int K, int Opad, int Oreal) {
        int tiles = (M / 16) * (Opad >> 6);
        gemm_wmma<<<dim3((unsigned)((tiles + 7) / 8)), 256, 0, stream>>>(A, Wt, bias, out,
                                                                         M, K, Opad, Oreal);
    };

    // norm1, projections
    ln_kernel<<<dim3(TOK), 128, 0, stream>>>(x, norm1_w, norm1_b, nullptr, y_h, 1, 0);
    gemm(y_h, qwh, q_b, q_f, TOK, 128, 128, 128);
    gemm(y_h, kvwh, kv_b, kv_f, TOK, 128, 256, 256);
    long bhn = (long)B_ * NH_ * N_;
    q_prep<<<grid1(bhn), TPB, 0, stream>>>(q_f, temp, qemb, qn_f, qs_f, qs_h);
    kvl_prep<<<grid1(bhn), TPB, 0, stream>>>(kv_f, kvl_f);
    local_logits<<<grid1(bhn), TPB, 0, stream>>>(qs_f, kvl_f, attnl);

    // pooled branch: sr(1x1) + gelu + LN -> kv projection -> k_p/v_p
    gemm(y_h, srwh, sr_b, s_f, TOK, 128, 128, 128);
    ln_kernel<<<dim3(TOK), 128, 0, stream>>>(s_f, pln_w, pln_b, nullptr, xp_h, 0, 1);
    gemm(xp_h, kvwh, kv_b, kvp_f, TOK, 128, 256, 256);
    kvp_prep<<<grid1(bhn), TPB, 0, stream>>>(kvp_f, kph, vpht);

    // flash pooled attention (LDS-staged) + local value branch
    flash_pool<<<dim3((unsigned)(B_ * NH_ * (N_ / 16 / FP_WAVES))), 32 * FP_WAVES, 0, stream>>>(
        qs_h, kph, vpht, attnl, xat_f, mbuf, zbuf);
    local_apply<<<grid1(bhn), TPB, 0, stream>>>(qn_f, ltok, lbias, attnl, kvl_f,
                                                mbuf, zbuf, xat_f, xat_h);

    // proj + residual
    gemm(xat_h, projwh, proj_b, proj_f, TOK, 128, 128, 128);
    add_res<<<grid1((long)TOK * C_), TPB, 0, stream>>>(x, proj_f, x1_f);

    // MLP (conv-GLU)
    ln_kernel<<<dim3(TOK), 128, 0, stream>>>(x1_f, norm2_w, norm2_b, nullptr, y2_h, 1, 0);
    gemm(y2_h, fc1wh, fc1_b, f1_f, TOK, 128, F1P, 682);
    dwglu<<<grid1((long)TOK * F2KP), TPB, 0, stream>>>(f1_f, dw_w, dw_b, g_h);
    gemm(g_h, fc2wh, fc2_b, mlp_f, TOK, F2KP, 128, 128);
    final_add<<<grid1((long)TOK * C_), TPB, 0, stream>>>(x1_f, mlp_f, (float*)d_out);
}
